// temporalGNN_25383256719676
// MI455X (gfx1250) — compile-verified, matrix operand order checked by reading, not run on hardware
//
#include <hip/hip_runtime.h>
#include <hip/hip_bf16.h>

// ---------------------------------------------------------------------------
// Temporal GNN: conv1x1(->bf16) -> GEMM(gcn_w) [WMMA bf16] -> GCN scatter
//   -> 3x GRU layer: wih panel pre-converted to bf16, input-gate GEMM
//      [WMMA bf16, 16x64 strip/wave, B panel staged in LDS per block]
//      + 256 sequential f32 GEMV steps -> final linear.
// The 4 waves of each GEMM block share one 64-col B strip: the B panel is
// double-buffered through LDS (4KB/chunk), cutting B global traffic 4x.
// ---------------------------------------------------------------------------

#define N_NODES   2048
#define SEQ       512
#define HIDDEN    256
#define N_FEAT    9
#define N_EDGES   65536
#define STEP_AHEAD 14
#define GRU_H     N_NODES        // 2048
#define GATES     (3 * GRU_H)    // 6144
#define T_STEPS   HIDDEN         // 256

typedef __attribute__((ext_vector_type(16))) __bf16 v16bf;
typedef __attribute__((ext_vector_type(8)))  __bf16 v8bf;
typedef __attribute__((ext_vector_type(8)))  float  v8f;

// ---------------- elementwise / setup kernels ----------------

__global__ void zero_kernel(float* __restrict__ p, int n) {
    int tid = blockIdx.x * blockDim.x + threadIdx.x;
    if (tid < n) p[tid] = 0.0f;
}

__global__ void cvt_bf16_kernel(const float* __restrict__ src,
                                __bf16* __restrict__ dst, int n) {
    int tid = blockIdx.x * blockDim.x + threadIdx.x;
    if (tid < n) dst[tid] = (__bf16)src[tid];
}

// Xbf[n*512+s] = bf16( sum_f cnn_w[f] * x[f,n,s] + cnn_b )
__global__ void conv_kernel(const float* __restrict__ x,
                            const float* __restrict__ cw,
                            const float* __restrict__ cb,
                            __bf16* __restrict__ Xbf) {
    int tid = blockIdx.x * blockDim.x + threadIdx.x;
    if (tid >= N_NODES * SEQ) return;
    float s = cb[0];
#pragma unroll
    for (int f = 0; f < N_FEAT; ++f)
        s += cw[f] * x[(size_t)f * N_NODES * SEQ + tid];
    Xbf[tid] = (__bf16)s;
}

// gwT_bf[n*512 + k] = bf16( gcn_w[k*256 + n] )
__global__ void transpose_bf_kernel(const float* __restrict__ w,
                                    __bf16* __restrict__ wT) {
    int tid = blockIdx.x * blockDim.x + threadIdx.x;  // SEQ * HIDDEN
    int n = tid & (HIDDEN - 1);
    int k = tid >> 8;
    wT[(size_t)n * SEQ + k] = (__bf16)w[(size_t)k * HIDDEN + n];
}

// ---------------- WMMA bf16 GEMM, B staged through LDS ----------------
// ISA 16-bit 16x32 fragment: per lane, two contiguous 8-element K spans:
// [k0+8*hi, +8) and [k0+16+8*hi, +8).
__device__ __forceinline__ v16bf load_frag_g(const __bf16* __restrict__ row,
                                             int k0, int hi) {
    v8bf s0 = *(const v8bf*)(row + k0 + (hi << 3));
    v8bf s1 = *(const v8bf*)(row + k0 + 16 + (hi << 3));
    v16bf f;
#pragma unroll
    for (int i = 0; i < 8; ++i) { f[i] = s0[i]; f[i + 8] = s1[i]; }
    return f;
}

// C(MxN) = A(MxK) @ Bt(NxK)^T (+bias[n]).
// Block = 4 waves; waves b*4..b*4+3 share tn (tilesM % 4 == 0), each owns one
// 16x64 strip. B panel (64 rows x 32 K bf16 = 4KB) double-buffered in LDS.
#define BPAD 40   // bf16 per LDS row (32 data + 8 pad); 80B = 5*16B, aligned
__global__ void wmma_gemm_bf16_lds(const __bf16* __restrict__ A,
                                   const __bf16* __restrict__ Bt,
                                   float* __restrict__ C,
                                   const float* __restrict__ bias,
                                   int M, int N, int K) {
    __shared__ __bf16 Bs[2][64 * BPAD];

    const int wid  = threadIdx.x >> 5;
    const int lane = threadIdx.x & 31;
    const int tilesM = M >> 4;
    const int tile = blockIdx.x * 4 + wid;       // grid sized exactly: no guard
    const int tm = tile % tilesM;
    const int tn = tile / tilesM;                // same for all 4 waves in block
    const int r  = lane & 15;
    const int hi = lane >> 4;

    // B staging: 128 threads, each copies 16 bf16 (row srow, half shalf)
    const int srow  = threadIdx.x >> 1;          // 0..63
    const int shalf = (threadIdx.x & 1) << 4;    // 0 or 16
    const __bf16* Bsrc = Bt + (size_t)(tn * 64 + srow) * K + shalf;

    const __bf16* Arow = A + (size_t)(tm * 16 + r) * K;

    v8f acc[4];
#pragma unroll
    for (int j = 0; j < 4; ++j) acc[j] = (v8f){};

    // prologue: stage chunk 0 into buffer 0
    *(v8bf*)&Bs[0][srow * BPAD + shalf]     = *(const v8bf*)(Bsrc + 0);
    *(v8bf*)&Bs[0][srow * BPAD + shalf + 8] = *(const v8bf*)(Bsrc + 8);

    int buf = 0;
    for (int k0 = 0; k0 < K; k0 += 32) {
        __syncthreads();   // staged buf ready; previous reads of buf^1 done
        if (k0 + 32 < K) {
            __builtin_prefetch(Bsrc + k0 + 288, 0, 0);   // global_prefetch_b8
            *(v8bf*)&Bs[buf ^ 1][srow * BPAD + shalf]     = *(const v8bf*)(Bsrc + k0 + 32);
            *(v8bf*)&Bs[buf ^ 1][srow * BPAD + shalf + 8] = *(const v8bf*)(Bsrc + k0 + 40);
        }
        v16bf a = load_frag_g(Arow, k0, hi);
        __builtin_prefetch(Arow + k0 + 256, 0, 0);
#pragma unroll
        for (int j = 0; j < 4; ++j) {
            const int prow = j * 16 + r;
            v8bf s0 = *(const v8bf*)&Bs[buf][prow * BPAD + (hi << 3)];
            v8bf s1 = *(const v8bf*)&Bs[buf][prow * BPAD + 16 + (hi << 3)];
            v16bf b;
#pragma unroll
            for (int i = 0; i < 8; ++i) { b[i] = s0[i]; b[i + 8] = s1[i]; }
            acc[j] = __builtin_amdgcn_wmma_f32_16x16x32_bf16(
                false, a, false, b, (short)0, acc[j], false, false);
        }
        buf ^= 1;
    }

#pragma unroll
    for (int j = 0; j < 4; ++j) {
        const int col  = tn * 64 + j * 16 + r;
        const float bv = bias ? bias[col] : 0.0f;
#pragma unroll
        for (int g = 0; g < 8; ++g) {
            // C layout: lane&15 -> N, VGPR g -> M = g + 8*hi
            C[(size_t)(tm * 16 + g + (hi << 3)) * N + col] = acc[j][g] + bv;
        }
    }
}

// ---------------- GCN aggregation ----------------

__global__ void deg_kernel(const int* __restrict__ ei,
                           const float* __restrict__ ew,
                           float* __restrict__ deg) {
    int e = blockIdx.x * blockDim.x + threadIdx.x;
    if (e >= N_EDGES) return;
    atomicAdd(&deg[ei[N_EDGES + e]], ew[e]);   // targets = row 1
}

__global__ void dinv_kernel(const float* __restrict__ deg,
                            float* __restrict__ dinv) {
    int n = blockIdx.x * blockDim.x + threadIdx.x;
    if (n >= N_NODES) return;
    float d = deg[n] + 1.0f;                   // + self-loop weight
    dinv[n] = rsqrtf(fmaxf(d, 1e-12f));
}

__global__ void scatter_kernel(const int* __restrict__ ei,
                               const float* __restrict__ ew,
                               const float* __restrict__ dinv,
                               const float* __restrict__ H0,
                               float* __restrict__ H1) {
    int tid = blockIdx.x * blockDim.x + threadIdx.x;  // N_EDGES * HIDDEN
    int c = tid & (HIDDEN - 1);
    int e = tid >> 8;
    int src = ei[e];
    int dst = ei[N_EDGES + e];
    float norm = dinv[src] * ew[e] * dinv[dst];
    atomicAdd(&H1[(size_t)dst * HIDDEN + c], norm * H0[(size_t)src * HIDDEN + c]);
}

// self-loop + bias + relu, write bf16 time-major seq[t=c][i=n] (GEMM A operand)
__global__ void relu_transpose_kernel(const float* __restrict__ H1,
                                      const float* __restrict__ H0,
                                      const float* __restrict__ dinv,
                                      const float* __restrict__ gb,
                                      __bf16* __restrict__ seq_bf) {
    int tid = blockIdx.x * blockDim.x + threadIdx.x;  // N_NODES * HIDDEN
    int c = tid & (HIDDEN - 1);
    int n = tid >> 8;
    float di = dinv[n];
    float v = H1[(size_t)n * HIDDEN + c] + di * di * H0[(size_t)n * HIDDEN + c] + gb[c];
    v = fmaxf(v, 0.0f);
    seq_bf[(size_t)c * N_NODES + n] = (__bf16)v;
}

// ---------------- GRU recurrence: one wave per hidden unit ----------------
__global__ void gru_step_kernel(const float* __restrict__ h_in,
                                float* __restrict__ h_out,
                                const float* __restrict__ gi,
                                const float* __restrict__ whh,
                                const float* __restrict__ bhh,
                                __bf16* __restrict__ ys_bf,
                                float* __restrict__ ys_f32, int t) {
    const int wid  = threadIdx.x >> 5;
    const int lane = threadIdx.x & 31;
    const int i = blockIdx.x * (blockDim.x >> 5) + wid;
    if (i >= GRU_H) return;

    const float4* h4 = (const float4*)h_in;
    const float4* wr = (const float4*)(whh + (size_t)i * GRU_H);
    const float4* wz = (const float4*)(whh + (size_t)(GRU_H + i) * GRU_H);
    const float4* wn = (const float4*)(whh + (size_t)(2 * GRU_H + i) * GRU_H);

    float dr = 0.f, dz = 0.f, dn = 0.f;
    for (int k = lane; k < GRU_H / 4; k += 32) {
        float4 hv = h4[k];
        float4 a = wr[k]; dr += hv.x * a.x + hv.y * a.y + hv.z * a.z + hv.w * a.w;
        float4 b = wz[k]; dz += hv.x * b.x + hv.y * b.y + hv.z * b.z + hv.w * b.w;
        float4 c = wn[k]; dn += hv.x * c.x + hv.y * c.y + hv.z * c.z + hv.w * c.w;
    }
#pragma unroll
    for (int off = 16; off > 0; off >>= 1) {
        dr += __shfl_xor(dr, off, 32);
        dz += __shfl_xor(dz, off, 32);
        dn += __shfl_xor(dn, off, 32);
    }
    if (lane == 0) {
        const float* git = gi + (size_t)t * GATES;
        float ghr = dr + bhh[i];
        float ghz = dz + bhh[GRU_H + i];
        float ghn = dn + bhh[2 * GRU_H + i];
        float rg = 1.0f / (1.0f + __expf(-(git[i] + ghr)));
        float zg = 1.0f / (1.0f + __expf(-(git[GRU_H + i] + ghz)));
        float ng = tanhf(git[2 * GRU_H + i] + rg * ghn);
        float hn = (1.0f - zg) * ng + zg * h_in[i];
        h_out[i] = hn;
        ys_bf[(size_t)t * GRU_H + i] = (__bf16)hn;   // next layer's GEMM A
        ys_f32[(size_t)t * GRU_H + i] = hn;          // final linear input
    }
}

// out[n][o] = sum_t seq[t][n] * lin_w[t][o] + lin_b[o]
__global__ void final_linear_kernel(const float* __restrict__ seq,
                                    const float* __restrict__ lw,
                                    const float* __restrict__ lb,
                                    float* __restrict__ out) {
    int tid = blockIdx.x * blockDim.x + threadIdx.x;
    if (tid >= N_NODES * STEP_AHEAD) return;
    int o = tid % STEP_AHEAD;
    int n = tid / STEP_AHEAD;
    float s = lb[o];
    for (int t = 0; t < T_STEPS; ++t)
        s += seq[(size_t)t * N_NODES + n] * lw[t * STEP_AHEAD + o];
    out[tid] = s;
}

// ---------------- host launch ----------------

extern "C" void kernel_launch(void* const* d_in, const int* in_sizes, int n_in,
                              void* d_out, int out_size, void* d_ws, size_t ws_size,
                              hipStream_t stream) {
    const float* x     = (const float*)d_in[0];
    const int*   ei    = (const int*)  d_in[1];
    const float* ew    = (const float*)d_in[2];
    const float* cnn_w = (const float*)d_in[3];
    const float* cnn_b = (const float*)d_in[4];
    const float* gcn_w = (const float*)d_in[5];
    const float* gcn_b = (const float*)d_in[6];
    const float* w_ih  = (const float*)d_in[7];
    const float* w_hh  = (const float*)d_in[8];
    const float* b_ih  = (const float*)d_in[9];
    const float* b_hh  = (const float*)d_in[10];
    const float* lin_w = (const float*)d_in[11];
    const float* lin_b = (const float*)d_in[12];

    // workspace layout (byte offsets, 256B-aligned)
    char* base = (char*)d_ws;
    size_t off = 0;
    auto take = [&](size_t bytes) { char* p = base + off; off = (off + bytes + 255) & ~(size_t)255; return p; };
    __bf16* Xbf   = (__bf16*)take((size_t)N_NODES * SEQ * 2);        //  2 MB
    __bf16* gwTbf = (__bf16*)take((size_t)HIDDEN * SEQ * 2);         //  0.25 MB
    float*  H0    = (float*) take((size_t)N_NODES * HIDDEN * 4);     //  2 MB
    float*  H1    = (float*) take((size_t)N_NODES * HIDDEN * 4);     //  2 MB
    float*  deg   = (float*) take((size_t)N_NODES * 4);
    float*  dinv  = (float*) take((size_t)N_NODES * 4);
    __bf16* seqAb = (__bf16*)take((size_t)T_STEPS * GRU_H * 2);      //  1 MB
    __bf16* seqBb = (__bf16*)take((size_t)T_STEPS * GRU_H * 2);      //  1 MB
    float*  seqF  = (float*) take((size_t)T_STEPS * GRU_H * 4);      //  2 MB
    float*  gi    = (float*) take((size_t)T_STEPS * GATES * 4);      //  6 MB
    __bf16* wbf   = (__bf16*)take((size_t)GATES * GRU_H * 2);        // 25.2 MB
    float*  hA    = (float*) take((size_t)GRU_H * 4);
    float*  hB    = (float*) take((size_t)GRU_H * 4);                // total ~42 MB

    // 1) conv 1x1 (writes bf16 A operand); transpose+convert gcn_w to (N x K) bf16
    conv_kernel<<<(N_NODES * SEQ + 255) / 256, 256, 0, stream>>>(x, cnn_w, cnn_b, Xbf);
    transpose_bf_kernel<<<(SEQ * HIDDEN) / 256, 256, 0, stream>>>(gcn_w, gwTbf);

    // 2) H0 = X @ gcn_w  (2048x512 * 512x256): 512 tiles -> 128 blocks x 4 waves
    wmma_gemm_bf16_lds<<<128, 128, 0, stream>>>(Xbf, gwTbf, H0, nullptr,
                                                N_NODES, HIDDEN, SEQ);

    // 3) GCN aggregation
    zero_kernel<<<(N_NODES * HIDDEN + 255) / 256, 256, 0, stream>>>(H1, N_NODES * HIDDEN);
    zero_kernel<<<(N_NODES + 255) / 256, 256, 0, stream>>>(deg, N_NODES);
    deg_kernel<<<N_EDGES / 256, 256, 0, stream>>>(ei, ew, deg);
    dinv_kernel<<<N_NODES / 256, 256, 0, stream>>>(deg, dinv);
    scatter_kernel<<<(N_EDGES * HIDDEN) / 256, 256, 0, stream>>>(ei, ew, dinv, H0, H1);
    relu_transpose_kernel<<<(N_NODES * HIDDEN) / 256, 256, 0, stream>>>(H1, H0, dinv, gcn_b, seqAb);

    // 4) GRU layers
    __bf16* cur = seqAb;
    __bf16* nxt = seqBb;
    for (int l = 0; l < 3; ++l) {
        // stage this layer's wih panel as bf16 (halves GEMM B traffic)
        cvt_bf16_kernel<<<((size_t)GATES * GRU_H + 255) / 256, 256, 0, stream>>>(
            w_ih + (size_t)l * GATES * GRU_H, wbf, GATES * GRU_H);
        // gi = cur @ wih.T + bih : (256x2048)*(2048x6144): 1536 tiles -> 384 blocks
        wmma_gemm_bf16_lds<<<384, 128, 0, stream>>>(cur, wbf, gi,
                                                    b_ih + (size_t)l * GATES,
                                                    T_STEPS, GATES, GRU_H);
        zero_kernel<<<GRU_H / 256, 256, 0, stream>>>(hA, GRU_H);
        for (int t = 0; t < T_STEPS; ++t) {
            const float* hin  = (t & 1) ? hB : hA;
            float*       hout = (t & 1) ? hA : hB;
            gru_step_kernel<<<512, 128, 0, stream>>>(hin, hout, gi,
                                                     w_hh + (size_t)l * GATES * GRU_H,
                                                     b_hh + (size_t)l * GATES,
                                                     nxt, seqF, t);
        }
        __bf16* tmp = cur; cur = nxt; nxt = tmp;
    }

    // 5) final linear: (2048 x 256) @ (256 x 14), reads last layer's f32 outputs
    final_linear_kernel<<<(N_NODES * STEP_AHEAD + 255) / 256, 256, 0, stream>>>(
        seqF, lin_w, lin_b, (float*)d_out);
}
